// Segger_24008867184711
// MI455X (gfx1250) — matrix operand order repaired and checked by reading, not compiled, not run on hardware
//
#include <hip/hip_runtime.h>
#include <hip/hip_bf16.h>

typedef __attribute__((ext_vector_type(16))) _Float16 v16h;
typedef __attribute__((ext_vector_type(8)))  float    v8f;

static constexpr int cN_TX  = 100000;
static constexpr int cN_BD  = 10000;
static constexpr int cE_NN  = 1600000;
static constexpr int cE_BEL = 300000;
static constexpr int cE_LBL = 500000;

// ---------------------------------------------------------------- helpers

__device__ __forceinline__ void wave_lds_sync() {
  // wave32 lanes are lockstep; LDS cross-lane visibility only needs DScnt==0.
  asm volatile("s_wait_dscnt 0" ::: "memory");
}

__device__ __forceinline__ v8f wmma16(v16h a, v16h b, v8f c) {
  return __builtin_amdgcn_wmma_f32_16x16x32_f16(false, a, false, b, (short)0, c,
                                                false, false);
}

__device__ __forceinline__ v8f prelu8(v8f x, float al) {
  v8f r;
#pragma unroll
  for (int i = 0; i < 8; ++i) r[i] = x[i] > 0.f ? x[i] : al * x[i];
  return r;
}

// A fragment (16xK tile, row-major f16 in LDS): lane = M (mod 16),
// halves 0..7 -> K base+{0..7 | 8..15}, halves 8..15 -> K base+{16..23 | 24..31}
__device__ __forceinline__ v16h load_a_frag(const _Float16* tile, int rowStride,
                                            int kbase, int lane) {
  const int m  = lane & 15;
  const int hi = (lane >> 4) & 1;
  const _Float16* p = tile + m * rowStride + kbase + hi * 8;
  union { v16h v; uint4 q[2]; } u;
  u.q[0] = *(const uint4*)p;
  u.q[1] = *(const uint4*)(p + 16);
  return u.v;
}

// B fragment from f32 row-major weights W[K][32]: lane = N, K = kbase + {hi*8 + h | 16 + hi*8 + h-8}
__device__ __forceinline__ v16h load_b_frag(const float* __restrict__ W,
                                            int kbase, int nh, int lane) {
  const int n  = nh * 16 + (lane & 15);
  const int hi = (lane >> 4) & 1;
  union { v16h v; _Float16 e[16]; } u;
#pragma unroll
  for (int hh = 0; hh < 16; ++hh) {
    const int k = kbase + (hh < 8 ? hi * 8 + hh : 16 + hi * 8 + (hh - 8));
    u.e[hh] = (_Float16)W[k * 32 + n];
  }
  return u.v;
}

template <int IN>
__device__ __forceinline__ void copy_row(_Float16* dstL, const _Float16* srcG) {
  const uint4* s = (const uint4*)srcG;
  uint4* d = (uint4*)dstL;
#pragma unroll
  for (int i = 0; i < IN / 8; ++i) d[i] = s[i];
}

// ---------------------------------------------------------------- kernels

__global__ __launch_bounds__(256) void embed_tx_kernel(
    const int* __restrict__ xtx, const float* __restrict__ emb,
    _Float16* __restrict__ out) {
  const int i = blockIdx.x * 256 + threadIdx.x;
  if (i >= cN_TX * 16) return;
  const int node = i >> 4, j = i & 15;
  out[i] = (_Float16)emb[xtx[node] * 16 + j];
}

__global__ __launch_bounds__(256) void init_bd_kernel(
    const float* __restrict__ xbd, const float* __restrict__ W,
    const float* __restrict__ B, _Float16* __restrict__ out) {
  __shared__ float sW[32 * 16];
  __shared__ float sB[16];
  for (int i = threadIdx.x; i < 32 * 16; i += 256) sW[i] = W[i];
  if (threadIdx.x < 16) sB[threadIdx.x] = B[threadIdx.x];
  __syncthreads();
  const int node = blockIdx.x * 256 + threadIdx.x;
  if (node >= cN_BD) return;
  float x[32];
#pragma unroll
  for (int k = 0; k < 32; ++k) x[k] = xbd[(size_t)node * 32 + k];
#pragma unroll
  for (int j = 0; j < 16; ++j) {
    float a = sB[j];
#pragma unroll
    for (int k = 0; k < 32; ++k) a += x[k] * sW[k * 16 + j];
    out[(size_t)node * 16 + j] = (_Float16)a;
  }
}

// Fused 2-layer edge MLP + atomic scatter-sum.  16 edges per wave per tile.
template <int IN>
__global__ __launch_bounds__(256) void edge_mlp_kernel(
    const _Float16* __restrict__ xdst, const _Float16* __restrict__ xsrc,
    const int* __restrict__ srcIdx, const int* __restrict__ dstIdx, int E,
    const float* __restrict__ W1, const float* __restrict__ B1,
    const float* __restrict__ A1p, const float* __restrict__ W2,
    const float* __restrict__ B2, const float* __restrict__ A2p,
    float* __restrict__ agg) {
  constexpr int K2 = 2 * IN;   // concat feature dim (32 or 64)
  constexpr int KC = K2 / 32;  // WMMA K-chunks for layer 1
  constexpr int WAVES = 8;
  __shared__ _Float16 sIn[WAVES][16 * K2];
  __shared__ _Float16 sMid[WAVES][16 * 32];
  __shared__ float    sOut[WAVES][16 * 32];
  __shared__ int      sDst[WAVES][16];
  __shared__ int      sVal[WAVES][16];

  const int lane = threadIdx.x & 31;
  const int wave = threadIdx.x >> 5;
  const float al1 = A1p[0], al2 = A2p[0];

  v16h Bw1[KC][2];
#pragma unroll
  for (int kc = 0; kc < KC; ++kc)
#pragma unroll
    for (int nh = 0; nh < 2; ++nh) Bw1[kc][nh] = load_b_frag(W1, kc * 32, nh, lane);
  v16h Bw2[2];
#pragma unroll
  for (int nh = 0; nh < 2; ++nh) Bw2[nh] = load_b_frag(W2, 0, nh, lane);
  float bias1[2], bias2[2];
#pragma unroll
  for (int nh = 0; nh < 2; ++nh) {
    const int n = nh * 16 + (lane & 15);
    bias1[nh] = B1[n];
    bias2[nh] = B2[n];
  }

  const int tiles = (E + 15) >> 4;
  const int gw = blockIdx.x * WAVES + wave;
  const int nw = gridDim.x * WAVES;
  for (int t = gw; t < tiles; t += nw) {
    const int e0 = t << 4;
    {  // gather: lanes 0..15 fetch dst half of row r, lanes 16..31 src half
      const int r = lane & 15;
      const int e = e0 + r;
      const bool v = (e < E);
      if (lane < 16) {
        const int d = v ? dstIdx[e] : 0;
        sDst[wave][r] = d;
        sVal[wave][r] = v ? 1 : 0;
        copy_row<IN>(&sIn[wave][r * K2], xdst + (size_t)d * IN);
      } else {
        const int s = v ? srcIdx[e] : 0;
        copy_row<IN>(&sIn[wave][r * K2 + IN], xsrc + (size_t)s * IN);
      }
    }
    wave_lds_sync();

    // layer 1: [16 x K2] x [K2 x 32] via WMMA, f32 acc, prelu, back to f16 LDS
    v16h a0 = load_a_frag(&sIn[wave][0], K2, 0, lane);
    v16h a1{};
    if constexpr (KC == 2) a1 = load_a_frag(&sIn[wave][0], K2, 32, lane);
#pragma unroll
    for (int nh = 0; nh < 2; ++nh) {
      v8f acc;
#pragma unroll
      for (int i = 0; i < 8; ++i) acc[i] = bias1[nh];
      acc = wmma16(a0, Bw1[0][nh], acc);
      if constexpr (KC == 2) acc = wmma16(a1, Bw1[1][nh], acc);
      acc = prelu8(acc, al1);
      const int n  = nh * 16 + (lane & 15);
      const int rb = (lane >> 4) << 3;
#pragma unroll
      for (int r2 = 0; r2 < 8; ++r2)
        sMid[wave][(rb + r2) * 32 + n] = (_Float16)acc[r2];
    }
    wave_lds_sync();

    // layer 2: [16 x 32] x [32 x 32]
    v16h a2 = load_a_frag(&sMid[wave][0], 32, 0, lane);
#pragma unroll
    for (int nh = 0; nh < 2; ++nh) {
      v8f acc;
#pragma unroll
      for (int i = 0; i < 8; ++i) acc[i] = bias2[nh];
      acc = wmma16(a2, Bw2[nh], acc);
      acc = prelu8(acc, al2);
      const int n  = nh * 16 + (lane & 15);
      const int rb = (lane >> 4) << 3;
#pragma unroll
      for (int r2 = 0; r2 < 8; ++r2)
        sOut[wave][(rb + r2) * 32 + n] = acc[r2];
    }
    wave_lds_sync();

    // scatter-sum: one coalesced row of 32 atomics per valid edge
#pragma unroll 4
    for (int r = 0; r < 16; ++r) {
      if (sVal[wave][r])
        atomicAdd(&agg[(size_t)sDst[wave][r] * 32 + lane],
                  sOut[wave][r * 32 + lane]);
    }
  }
}

// Node update: concat[x, agg] -> n1 -> prelu -> n2 -> prelu ; emit pre-BN f32
// plus block-reduced per-column sum / sum-of-squares for BatchNorm stats.
template <int IN>
__global__ __launch_bounds__(256) void node_mlp_kernel(
    const _Float16* __restrict__ x, const float* __restrict__ agg, int N,
    const float* __restrict__ W1, const float* __restrict__ B1,
    const float* __restrict__ A1p, const float* __restrict__ W2,
    const float* __restrict__ B2, const float* __restrict__ A2p,
    float* __restrict__ tmp, float* __restrict__ gsum, float* __restrict__ gsq) {
  constexpr int KH = IN + 32;
  __shared__ float sW1[KH * 32];
  __shared__ float sW2[32 * 32];
  __shared__ float sB1[32], sB2[32], sSum[32], sSq[32];
  for (int i = threadIdx.x; i < KH * 32; i += 256) sW1[i] = W1[i];
  for (int i = threadIdx.x; i < 32 * 32; i += 256) sW2[i] = W2[i];
  if (threadIdx.x < 32) {
    sB1[threadIdx.x] = B1[threadIdx.x];
    sB2[threadIdx.x] = B2[threadIdx.x];
    sSum[threadIdx.x] = 0.f;
    sSq[threadIdx.x] = 0.f;
  }
  __syncthreads();
  const float al1 = A1p[0], al2 = A2p[0];
  const int node = blockIdx.x * 256 + threadIdx.x;
  if (node < N) {
    float h[KH];
#pragma unroll
    for (int k = 0; k < IN; ++k) h[k] = (float)x[(size_t)node * IN + k];
#pragma unroll
    for (int k = 0; k < 32; ++k) h[IN + k] = agg[(size_t)node * 32 + k];
    float u[32];
#pragma unroll
    for (int j = 0; j < 32; ++j) {
      float a = sB1[j];
#pragma unroll
      for (int k = 0; k < KH; ++k) a += h[k] * sW1[k * 32 + j];
      u[j] = a > 0.f ? a : al1 * a;
    }
#pragma unroll
    for (int j = 0; j < 32; ++j) {
      float a = sB2[j];
#pragma unroll
      for (int k = 0; k < 32; ++k) a += u[k] * sW2[k * 32 + j];
      const float y = a > 0.f ? a : al2 * a;
      tmp[(size_t)node * 32 + j] = y;
      atomicAdd(&sSum[j], y);
      atomicAdd(&sSq[j], y * y);
    }
  }
  __syncthreads();
  if (threadIdx.x < 32) {
    atomicAdd(&gsum[threadIdx.x], sSum[threadIdx.x]);
    atomicAdd(&gsq[threadIdx.x], sSq[threadIdx.x]);
  }
}

__global__ __launch_bounds__(256) void bn_apply_kernel(
    const float* __restrict__ tmp, const float* __restrict__ gsum,
    const float* __restrict__ gsq, const float* __restrict__ g,
    const float* __restrict__ be, int N, _Float16* __restrict__ out) {
  const int i = blockIdx.x * 256 + threadIdx.x;
  if (i >= N * 32) return;
  const int j = i & 31;
  const float m = gsum[j] / (float)N;
  const float v = gsq[j] / (float)N - m * m;
  const float r = rsqrtf(v + 1e-5f);
  out[i] = (_Float16)((tmp[i] - m) * r * g[j] + be[j]);
}

__global__ __launch_bounds__(256) void mlp_head_kernel(
    const _Float16* __restrict__ x, int N, const float* __restrict__ W1,
    const float* __restrict__ B1, const float* __restrict__ Ap,
    const float* __restrict__ W2, const float* __restrict__ B2,
    float* __restrict__ out) {
  __shared__ float sW1[32 * 32], sW2[32 * 32], sB1[32], sB2[32];
  for (int i = threadIdx.x; i < 32 * 32; i += 256) {
    sW1[i] = W1[i];
    sW2[i] = W2[i];
  }
  if (threadIdx.x < 32) {
    sB1[threadIdx.x] = B1[threadIdx.x];
    sB2[threadIdx.x] = B2[threadIdx.x];
  }
  __syncthreads();
  const int node = blockIdx.x * 256 + threadIdx.x;
  if (node >= N) return;
  const float al = Ap[0];
  float h[32];
#pragma unroll
  for (int k = 0; k < 32; ++k) h[k] = (float)x[(size_t)node * 32 + k];
  float u[32];
#pragma unroll
  for (int j = 0; j < 32; ++j) {
    float a = sB1[j];
#pragma unroll
    for (int k = 0; k < 32; ++k) a += h[k] * sW1[k * 32 + j];
    u[j] = a > 0.f ? a : al * a;
  }
#pragma unroll
  for (int j = 0; j < 32; ++j) {
    float a = sB2[j];
#pragma unroll
    for (int k = 0; k < 32; ++k) a += u[k] * sW2[k * 32 + j];
    out[(size_t)node * 32 + j] = a;
  }
}

__global__ __launch_bounds__(256) void edge_head_kernel(
    const float* __restrict__ txo, const float* __restrict__ bdo,
    const int* __restrict__ etx, const int* __restrict__ ebd,
    const float* __restrict__ A0p, const float* __restrict__ A1p,
    const float* __restrict__ W1, const float* __restrict__ B1,
    const float* __restrict__ W2, const float* __restrict__ B2,
    float* __restrict__ out) {
  __shared__ float sW1[64 * 32], sB1[32], sW2[32];
  for (int i = threadIdx.x; i < 64 * 32; i += 256) sW1[i] = W1[i];
  if (threadIdx.x < 32) {
    sB1[threadIdx.x] = B1[threadIdx.x];
    sW2[threadIdx.x] = W2[threadIdx.x];
  }
  __syncthreads();
  const int e = blockIdx.x * 256 + threadIdx.x;
  if (e >= cE_LBL) return;
  const float a0 = A0p[0], a1 = A1p[0];
  const int it = etx[e], ib = ebd[e];
  float h[64];
#pragma unroll
  for (int k = 0; k < 32; ++k) {
    const float v = txo[(size_t)it * 32 + k];
    h[k] = v > 0.f ? v : a0 * v;
  }
#pragma unroll
  for (int k = 0; k < 32; ++k) {
    const float v = bdo[(size_t)ib * 32 + k];
    h[32 + k] = v > 0.f ? v : a0 * v;
  }
  float u[32];
#pragma unroll
  for (int j = 0; j < 32; ++j) {
    float a = sB1[j];
#pragma unroll
    for (int k = 0; k < 64; ++k) a += h[k] * sW1[k * 32 + j];
    u[j] = a > 0.f ? a : a1 * a;
  }
  float o = B2[0];
#pragma unroll
  for (int k = 0; k < 32; ++k) o += u[k] * sW2[k];
  out[e] = o;
}

// ---------------------------------------------------------------- host

struct Lin { const float *W, *b; };
struct Blk {
  const float *a, *be, *e1W, *e1b, *e2W, *e2b, *ea1, *ea2, *g, *n1W, *n1b,
      *n2W, *n2b, *na1;
};

extern "C" void kernel_launch(void* const* d_in, const int* in_sizes, int n_in,
                              void* d_out, int out_size, void* d_ws,
                              size_t ws_size, hipStream_t stream) {
  (void)in_sizes; (void)n_in; (void)out_size; (void)ws_size;

  const int*   x_tx    = (const int*)d_in[0];
  const float* x_bd    = (const float*)d_in[1];
  const int*   ei_nn   = (const int*)d_in[2];
  const int*   src_nn  = ei_nn;
  const int*   dst_nn  = ei_nn + cE_NN;
  const int*   src_bel = (const int*)d_in[3];
  const int*   dst_bel = (const int*)d_in[4];
  const int*   eli_tx  = (const int*)d_in[5];
  const int*   eli_bd  = (const int*)d_in[6];

  // params flattened jax-style: nested dicts with sorted keys.
  int pi = 7;
  auto P = [&]() { return (const float*)d_in[pi++]; };
  auto readBlk = [&]() {
    Blk b;
    b.a = P(); b.be = P(); b.e1W = P(); b.e1b = P(); b.e2W = P(); b.e2b = P();
    b.ea1 = P(); b.ea2 = P(); b.g = P(); b.n1W = P(); b.n1b = P();
    b.n2W = P(); b.n2b = P(); b.na1 = P();
    return b;
  };
  Blk c1b = readBlk();   // c1_bel
  Blk c1n = readBlk();   // c1_nn
  Blk c2b = readBlk();   // c2_bel
  Blk c2n = readBlk();   // c2_nn
  const float* emb_tx = P();
  const float *em_a0 = P(), *em_a1 = P(), *em_l1W = P(), *em_l1b = P(),
              *em_l2W = P(), *em_l2b = P();              // emlp
  const float *ib_W = P(), *ib_b = P();                  // init_bd
  const float *mb_a = P(), *mb_l1W = P(), *mb_l1b = P(), *mb_l2W = P(),
              *mb_l2b = P();                             // mlp_bd
  const float *mt_a = P(), *mt_l1W = P(), *mt_l1b = P(), *mt_l2W = P(),
              *mt_l2b = P();                             // mlp_tx

  // workspace carve (256B aligned)
  char* ws = (char*)d_ws;
  size_t off = 0;
  auto carve = [&](size_t bytes) {
    void* p = ws + off;
    off = (off + bytes + 255) & ~(size_t)255;
    return p;
  };
  _Float16* hTx  = (_Float16*)carve((size_t)cN_TX * 16 * 2);
  _Float16* hBd  = (_Float16*)carve((size_t)cN_BD * 16 * 2);
  _Float16* fT1  = (_Float16*)carve((size_t)cN_TX * 32 * 2);
  _Float16* fB1  = (_Float16*)carve((size_t)cN_BD * 32 * 2);
  _Float16* fT2  = (_Float16*)carve((size_t)cN_TX * 32 * 2);
  _Float16* fB2  = (_Float16*)carve((size_t)cN_BD * 32 * 2);
  float* aggTx   = (float*)carve((size_t)cN_TX * 32 * 4);
  float* aggBd   = (float*)carve((size_t)cN_BD * 32 * 4);
  float* tmpTx   = (float*)carve((size_t)cN_TX * 32 * 4);
  float* tmpBd   = (float*)carve((size_t)cN_BD * 32 * 4);
  float* stats   = (float*)carve(128 * 4);
  float* sumTx = stats, *sqTx = stats + 32, *sumBd = stats + 64,
        *sqBd = stats + 96;

  float* outTx = (float*)d_out;
  float* outBd = outTx + (size_t)cN_TX * 32;
  float* outEx = outBd + (size_t)cN_BD * 32;

  auto cdiv = [](int a, int b) { return (a + b - 1) / b; };

  // ---- init features
  embed_tx_kernel<<<cdiv(cN_TX * 16, 256), 256, 0, stream>>>(x_tx, emb_tx, hTx);
  init_bd_kernel<<<cdiv(cN_BD, 256), 256, 0, stream>>>(x_bd, ib_W, ib_b, hBd);

  // ---- layer 1 (IN=16)
  hipMemsetAsync(aggTx, 0, (size_t)cN_TX * 32 * 4, stream);
  edge_mlp_kernel<16><<<1024, 256, 0, stream>>>(
      hTx, hTx, src_nn, dst_nn, cE_NN, c1n.e1W, c1n.e1b, c1n.ea1, c1n.e2W,
      c1n.e2b, c1n.ea2, aggTx);
  hipMemsetAsync(aggBd, 0, (size_t)cN_BD * 32 * 4, stream);
  edge_mlp_kernel<16><<<256, 256, 0, stream>>>(
      hBd, hTx, src_bel, dst_bel, cE_BEL, c1b.e1W, c1b.e1b, c1b.ea1, c1b.e2W,
      c1b.e2b, c1b.ea2, aggBd);
  hipMemsetAsync(stats, 0, 128 * 4, stream);
  node_mlp_kernel<16><<<cdiv(cN_TX, 256), 256, 0, stream>>>(
      hTx, aggTx, cN_TX, c1n.n1W, c1n.n1b, c1n.na1, c1n.n2W, c1n.n2b, c1n.a,
      tmpTx, sumTx, sqTx);
  node_mlp_kernel<16><<<cdiv(cN_BD, 256), 256, 0, stream>>>(
      hBd, aggBd, cN_BD, c1b.n1W, c1b.n1b, c1b.na1, c1b.n2W, c1b.n2b, c1b.a,
      tmpBd, sumBd, sqBd);
  bn_apply_kernel<<<cdiv(cN_TX * 32, 256), 256, 0, stream>>>(
      tmpTx, sumTx, sqTx, c1n.g, c1n.be, cN_TX, fT1);
  bn_apply_kernel<<<cdiv(cN_BD * 32, 256), 256, 0, stream>>>(
      tmpBd, sumBd, sqBd, c1b.g, c1b.be, cN_BD, fB1);

  // ---- layer 2 (IN=32)
  hipMemsetAsync(aggTx, 0, (size_t)cN_TX * 32 * 4, stream);
  edge_mlp_kernel<32><<<1024, 256, 0, stream>>>(
      fT1, fT1, src_nn, dst_nn, cE_NN, c2n.e1W, c2n.e1b, c2n.ea1, c2n.e2W,
      c2n.e2b, c2n.ea2, aggTx);
  hipMemsetAsync(aggBd, 0, (size_t)cN_BD * 32 * 4, stream);
  edge_mlp_kernel<32><<<256, 256, 0, stream>>>(
      fB1, fT1, src_bel, dst_bel, cE_BEL, c2b.e1W, c2b.e1b, c2b.ea1, c2b.e2W,
      c2b.e2b, c2b.ea2, aggBd);
  hipMemsetAsync(stats, 0, 128 * 4, stream);
  node_mlp_kernel<32><<<cdiv(cN_TX, 256), 256, 0, stream>>>(
      fT1, aggTx, cN_TX, c2n.n1W, c2n.n1b, c2n.na1, c2n.n2W, c2n.n2b, c2n.a,
      tmpTx, sumTx, sqTx);
  node_mlp_kernel<32><<<cdiv(cN_BD, 256), 256, 0, stream>>>(
      fB1, aggBd, cN_BD, c2b.n1W, c2b.n1b, c2b.na1, c2b.n2W, c2b.n2b, c2b.a,
      tmpBd, sumBd, sqBd);
  bn_apply_kernel<<<cdiv(cN_TX * 32, 256), 256, 0, stream>>>(
      tmpTx, sumTx, sqTx, c2n.g, c2n.be, cN_TX, fT2);
  bn_apply_kernel<<<cdiv(cN_BD * 32, 256), 256, 0, stream>>>(
      tmpBd, sumBd, sqBd, c2b.g, c2b.be, cN_BD, fB2);

  // ---- output heads
  mlp_head_kernel<<<cdiv(cN_TX, 256), 256, 0, stream>>>(
      fT2, cN_TX, mt_l1W, mt_l1b, mt_a, mt_l2W, mt_l2b, outTx);
  mlp_head_kernel<<<cdiv(cN_BD, 256), 256, 0, stream>>>(
      fB2, cN_BD, mb_l1W, mb_l1b, mb_a, mb_l2W, mb_l2b, outBd);
  edge_head_kernel<<<cdiv(cE_LBL, 256), 256, 0, stream>>>(
      outTx, outBd, eli_tx, eli_bd, em_a0, em_a1, em_l1W, em_l1b, em_l2W,
      em_l2b, outEx);
}
